// RNN_71923522339087
// MI455X (gfx1250) — compile-verified, matrix operand order checked by reading, not run on hardware
//
#include <hip/hip_runtime.h>
#include <hip/hip_fp16.h>
#include <math.h>

// CDNA5 WMMA vector types
typedef __attribute__((ext_vector_type(16))) _Float16 v16h;
typedef __attribute__((ext_vector_type(8)))  float    v8f;

#define BATCH   512
#define HDIM    256
#define ROWS    16    // batch rows per block (WMMA M)
#define HSTRIDE 264   // padded LDS row stride in halves: 132 dwords -> rows 4 banks apart

// Branch-free tanh: clamp -> v_exp_f32 -> v_rcp_f32.  ~6 VALU ops, no EXEC branches.
// tanh(x) = 1 - 2/(e^{2x}+1);  e^{2x} = exp2(x * 2*log2(e)).  |x|>10 saturates exactly.
__device__ __forceinline__ float fast_tanh(float v) {
    float xc = fminf(fmaxf(v, -10.f), 10.f);
    float e  = __builtin_amdgcn_exp2f(xc * 2.885390082f);
    return 1.f - 2.f * __builtin_amdgcn_rcpf(e + 1.f);
}

// One block = 16 batch rows, 256 threads = 8 waves; wave w owns output cols [32w, 32w+32).
// U is converted to f16 fragments ONCE and pinned in VGPRs for the entire T-step loop.
// h ping-pongs through LDS in f16.  Recurrence: h = tanh(x_t * W + b + h @ U).
__global__ __launch_bounds__(256)
void rnn_wmma_kernel(const float* __restrict__ x,    // [B, T] (innermost dim D==1 collapsed)
                     const float* __restrict__ W,    // [H]
                     const float* __restrict__ U,    // [H, H] row-major
                     const float* __restrict__ bias, // [H]
                     float* __restrict__ hout,       // [B, H] last hidden state (f32)
                     int T)
{
    __shared__ _Float16 hbuf[2][ROWS][HSTRIDE];
    __shared__ float    xc[2][ROWS];

    const int tid   = threadIdx.x;
    const int wave  = tid >> 5;
    const int lane  = tid & 31;
    const int l16   = lane & 15;
    const int hi    = lane >> 4;   // half-wave select (K-run / M-run split)
    const int mbase = hi * 8;      // C/D: VGPR i holds M = i + 8*hi, N = lane&15
    const int b0    = blockIdx.x * ROWS;

    // ---- Load U fragments (B-matrix, 32x16 K-major slices), f32 -> f16, keep in VGPRs ----
    v16h uf[2][8];
    #pragma unroll
    for (int tt = 0; tt < 2; ++tt) {
        const int n = wave * 32 + tt * 16 + l16;
        #pragma unroll
        for (int ks = 0; ks < 8; ++ks) {
            const int kb = ks * 32 + hi * 8;
            v16h b;
            #pragma unroll
            for (int i = 0; i < 8; ++i) {
                b[i]     = (_Float16)U[(kb + i)      * HDIM + n];
                b[8 + i] = (_Float16)U[(kb + 16 + i) * HDIM + n];
            }
            uf[tt][ks] = b;
        }
    }

    // Per-lane input-projection constants for this wave's two column tiles
    float wv[2], bv[2];
    #pragma unroll
    for (int tt = 0; tt < 2; ++tt) {
        const int n = wave * 32 + tt * 16 + l16;
        wv[tt] = W[n];
        bv[tt] = bias[n];
    }

    // h0 = 0 (buffer 0); preload x column for t=0
    for (int i = tid; i < ROWS * HSTRIDE; i += 256)
        hbuf[0][0][i] = (_Float16)0.f;
    if (tid < ROWS) xc[0][tid] = x[(b0 + tid) * T];
    __syncthreads();

    for (int t = 0; t < T; ++t) {
        const int cur = t & 1, nxt = cur ^ 1;

        // Prefetch next x column into the other buffer (covered by the end barrier)
        if (t + 1 < T && tid < ROWS) xc[nxt][tid] = x[(b0 + tid) * T + t + 1];

        // Prefetch ALL A fragments for this step: 16x ds_load_b128 issued up front
        // so the 16-WMMA chain below runs against resident registers.
        v16h a[8];
        #pragma unroll
        for (int ks = 0; ks < 8; ++ks) {
            const _Float16* abase = &hbuf[cur][l16][ks * 32 + hi * 8];
            #pragma unroll
            for (int i = 0; i < 8; ++i) { a[ks][i] = abase[i]; a[ks][8 + i] = abase[16 + i]; }
        }

        // C preload: acc = x_t * W + b  (folds input projection into the WMMA C operand)
        v8f acc[2];
        #pragma unroll
        for (int tt = 0; tt < 2; ++tt) {
            #pragma unroll
            for (int i = 0; i < 8; ++i)
                acc[tt][i] = xc[cur][mbase + i] * wv[tt] + bv[tt];
        }

        // acc += h_cur @ U : K=256 as 8 chained v_wmma_f32_16x16x32_f16 per tile
        #pragma unroll
        for (int ks = 0; ks < 8; ++ks) {
            acc[0] = __builtin_amdgcn_wmma_f32_16x16x32_f16(false, a[ks], false, uf[0][ks],
                                                            (short)0, acc[0], false, false);
            acc[1] = __builtin_amdgcn_wmma_f32_16x16x32_f16(false, a[ks], false, uf[1][ks],
                                                            (short)0, acc[1], false, false);
        }

        // h_next = tanh(acc) -> LDS (f16); final step also spilled to global in f32
        #pragma unroll
        for (int tt = 0; tt < 2; ++tt) {
            const int n = wave * 32 + tt * 16 + l16;
            #pragma unroll
            for (int i = 0; i < 8; ++i) {
                const float hv = fast_tanh(acc[tt][i]);
                hbuf[nxt][mbase + i][n] = (_Float16)hv;
                if (t == T - 1) hout[(b0 + mbase + i) * HDIM + n] = hv;
            }
        }
        __syncthreads();
    }
}

// Tiny MLP head: 256->128->64->32->9 with ReLU between.  One block per batch row.
__global__ __launch_bounds__(256)
void mlp_head_kernel(const float* __restrict__ h,
                     const float* __restrict__ Wd1, const float* __restrict__ bd1,
                     const float* __restrict__ Wd2, const float* __restrict__ bd2,
                     const float* __restrict__ Wd3, const float* __restrict__ bd3,
                     const float* __restrict__ Wd4, const float* __restrict__ bd4,
                     float* __restrict__ out)
{
    __shared__ float s0[256], s1[128], s2[64], s3[32];
    const int tid = threadIdx.x;
    const int row = blockIdx.x;

    s0[tid] = h[row * 256 + tid];
    __syncthreads();

    if (tid < 128) {
        float s = bd1[tid];
        for (int k = 0; k < 256; ++k) s += s0[k] * Wd1[k * 128 + tid];
        s1[tid] = fmaxf(s, 0.f);
    }
    __syncthreads();
    if (tid < 64) {
        float s = bd2[tid];
        for (int k = 0; k < 128; ++k) s += s1[k] * Wd2[k * 64 + tid];
        s2[tid] = fmaxf(s, 0.f);
    }
    __syncthreads();
    if (tid < 32) {
        float s = bd3[tid];
        for (int k = 0; k < 64; ++k) s += s2[k] * Wd3[k * 32 + tid];
        s3[tid] = fmaxf(s, 0.f);
    }
    __syncthreads();
    if (tid < 9) {
        float s = bd4[tid];
        for (int k = 0; k < 32; ++k) s += s3[k] * Wd4[k * 9 + tid];
        out[row * 9 + tid] = s;
    }
}

extern "C" void kernel_launch(void* const* d_in, const int* in_sizes, int n_in,
                              void* d_out, int out_size, void* d_ws, size_t ws_size,
                              hipStream_t stream) {
    const float* x   = (const float*)d_in[0];   // [512,1024,1]
    const float* W1  = (const float*)d_in[1];   // [1,256]
    const float* U1  = (const float*)d_in[2];   // [256,256]
    const float* b1  = (const float*)d_in[3];   // [256]
    const float* W2  = (const float*)d_in[4];   // [1,256]
    const float* U2  = (const float*)d_in[5];   // [256,256]
    const float* b2  = (const float*)d_in[6];   // [256]
    const float* Wd1 = (const float*)d_in[7];
    const float* bd1 = (const float*)d_in[8];
    const float* Wd2 = (const float*)d_in[9];
    const float* bd2 = (const float*)d_in[10];
    const float* Wd3 = (const float*)d_in[11];
    const float* bd3 = (const float*)d_in[12];
    const float* Wd4 = (const float*)d_in[13];
    const float* bd4 = (const float*)d_in[14];
    float* out = (float*)d_out;

    float* h1 = (float*)d_ws;               // [512,256]
    float* h2 = h1 + BATCH * HDIM;          // [512,256]

    // RNN layer 1: T=1024 over x
    rnn_wmma_kernel<<<BATCH / ROWS, 256, 0, stream>>>(x, W1, U1, b1, h1, 1024);
    // RNN layer 2: the 256 features of h1 are the sequence, T=256
    rnn_wmma_kernel<<<BATCH / ROWS, 256, 0, stream>>>(h1, W2, U2, b2, h2, 256);
    // Dense head
    mlp_head_kernel<<<BATCH, 256, 0, stream>>>(h2, Wd1, bd1, Wd2, bd2, Wd3, bd3,
                                               Wd4, bd4, out);
}
